// PinSageConv_67104569032741
// MI455X (gfx1250) — matrix operand order repaired and checked by reading, not compiled, not run on hardware
//
#include <hip/hip_runtime.h>
#include <hip/hip_bf16.h>

typedef _Float16 v16h __attribute__((ext_vector_type(16)));
typedef _Float16 v8h  __attribute__((ext_vector_type(8)));
typedef _Float16 v4h  __attribute__((ext_vector_type(4)));
typedef float    v8f  __attribute__((ext_vector_type(8)));

#define NUM_NODES 20000
#define T_NB      50
#define IN_F      256
#define CAT_F     512
#define OUT_F     256

// LDS pitches (f16 elems), padded +16 to rotate banks across rows
#define QP 272
#define AP 272
#define WP 528

__device__ __forceinline__ v16h cat8(v8h lo, v8h hi) {
  return __builtin_shufflevector(lo, hi, 0,1,2,3,4,5,6,7,8,9,10,11,12,13,14,15);
}

// ---------------- Kernel 1: per-neighbor Q GEMM + ReLU + weighted mean ----
// grid-stride over nodes; one node per WG iteration; 512 threads = 16 waves.
__global__ __launch_bounds__(512)
void k1_aggregate(const float* __restrict__ h,
                  const int*   __restrict__ nodeset,
                  const int*   __restrict__ nb_nodes,
                  const float* __restrict__ nb_weights,
                  const float* __restrict__ Qw,
                  const float* __restrict__ Qb,
                  _Float16*    __restrict__ concatOut) {
  __shared__ _Float16 Qlds[256 * QP];   // 139264 B
  __shared__ _Float16 Alds[64 * AP];    //  34816 B
  __shared__ float    wlds[64];
  __shared__ float    qblds[256];
  __shared__ float    agg[256];
  __shared__ float    invw;

  const int tid   = threadIdx.x;
  const int lane  = tid & 31;
  const int wave  = tid >> 5;
  const int laneN = lane & 15;       // column / M-row index within tile
  const int laneH = lane >> 4;       // lane-half selector

  // Stage Q_w (f32 -> f16) into LDS once per workgroup.
  for (int it = 0; it < 32; ++it) {
    int flat = it * 512 + tid;               // 16384 float4 total
    int row  = flat >> 6;
    int c4   = (flat & 63) << 2;
    float4 v = *(const float4*)(Qw + row * 256 + c4);
    v4h p; p.x = (_Float16)v.x; p.y = (_Float16)v.y;
           p.z = (_Float16)v.z; p.w = (_Float16)v.w;
    *(v4h*)&Qlds[row * QP + c4] = p;
  }
  if (tid < 256) qblds[tid] = Qb[tid];
  __syncthreads();

  const int mTile  = wave & 3;               // 4 M tiles (64 rows)
  const int nBase0 = (wave >> 2) * 4;        // 4 N tiles per wave (16 total)

  for (int node = blockIdx.x; node < NUM_NODES; node += gridDim.x) {
    // neighbor weights + 1/sum (wave 0 only; visible after barrier)
    if (wave == 0) {
      float w0 = (lane < T_NB) ? nb_weights[node * T_NB + lane] : 0.f;
      float w1 = (lane + 32 < T_NB) ? nb_weights[node * T_NB + lane + 32] : 0.f;
      wlds[lane] = w0; wlds[lane + 32] = w1;
      float s = w0 + w1;
      for (int off = 16; off > 0; off >>= 1) s += __shfl_xor(s, off, 32);
      if (lane == 0) invw = 1.f / s;
    }
    if (tid < 256) agg[tid] = 0.f;

    // Gather 50 neighbor rows of h -> f16 LDS tile (rows >= 50 left stale; masked later)
    for (int it = 0; it < 7; ++it) {
      int flat = it * 512 + tid;             // 3200 float4 total
      if (flat < 3200) {
        int row = flat >> 6;
        int c4  = (flat & 63) << 2;
        int src = nb_nodes[node * T_NB + row];
        float4 v = *(const float4*)(h + (long)src * IN_F + c4);
        v4h p; p.x = (_Float16)v.x; p.y = (_Float16)v.y;
               p.z = (_Float16)v.z; p.w = (_Float16)v.w;
        *(v4h*)&Alds[row * AP + c4] = p;
      }
    }
    __syncthreads();

    // D = A(64x256) x Qw^T(256x256): per wave 4 (16x16) tiles, K-loop of 8
    v8f acc[4] = {};
    const int arow = (mTile * 16 + laneN) * AP;
    #pragma unroll
    for (int kb = 0; kb < 256; kb += 32) {
      v8h alo = *(const v8h*)&Alds[arow + kb + laneH * 16];
      v8h ahi = *(const v8h*)&Alds[arow + kb + laneH * 16 + 8];
      v16h a  = cat8(alo, ahi);
      #pragma unroll
      for (int j = 0; j < 4; ++j) {
        const v16h b = *(const v16h*)&Qlds[((nBase0 + j) * 16 + laneN) * QP + kb + laneH * 16];
        acc[j] = __builtin_amdgcn_wmma_f32_16x16x32_f16(
                     false, a, false, b, (short)0, acc[j], false, false);
      }
    }

    // bias + ReLU + weighted partial sum over this wave's 16 rows -> LDS atomic
    const int rowBase = mTile * 16 + laneH * 8;
    #pragma unroll
    for (int j = 0; j < 4; ++j) {
      int col  = (nBase0 + j) * 16 + laneN;
      float qb = qblds[col];
      float s  = 0.f;
      #pragma unroll
      for (int r = 0; r < 8; ++r) {
        int row = rowBase + r;
        float v = acc[j][r] + qb;
        v = fmaxf(v, 0.f);
        if (row < T_NB) s += wlds[row] * v;
      }
      atomicAdd(&agg[col], s);
    }
    __syncthreads();

    // emit concat row: [ h[nodeset[node]] | agg/sum(w) ] as f16
    if (tid < 256) {
      int col  = tid;
      float av = agg[col] * invw;
      int  self = nodeset[node];
      float hv = h[(long)self * IN_F + col];
      concatOut[(long)node * CAT_F + col]         = (_Float16)hv;
      concatOut[(long)node * CAT_F + IN_F + col]  = (_Float16)av;
    }
    __syncthreads();   // protect agg/wlds/Alds before next iteration
  }
}

// ---------------- Kernel 2: concat x W^T + bias + ReLU + L2 normalize -----
__global__ __launch_bounds__(512)
void k2_output(const _Float16* __restrict__ concatIn,
               const float*    __restrict__ Ww,
               const float*    __restrict__ Wb,
               float*          __restrict__ out) {
  __shared__ _Float16 Wlds[256 * WP];   // 270336 B
  __shared__ _Float16 Alds[32 * WP];    //  33792 B (aliased as f32 out tile 32x256)
  __shared__ float    blds[256];

  const int tid   = threadIdx.x;
  const int lane  = tid & 31;
  const int wave  = tid >> 5;
  const int laneN = lane & 15;
  const int laneH = lane >> 4;

  // Stage W_w (f32 -> f16) into LDS once per workgroup.
  for (int it = 0; it < 64; ++it) {
    int flat = it * 512 + tid;            // 32768 float4 total
    int row  = flat >> 7;                 // 128 float4 per 512-wide row
    int c4   = (flat & 127) << 2;
    float4 v = *(const float4*)(Ww + row * CAT_F + c4);
    v4h p; p.x = (_Float16)v.x; p.y = (_Float16)v.y;
           p.z = (_Float16)v.z; p.w = (_Float16)v.w;
    *(v4h*)&Wlds[row * WP + c4] = p;
  }
  if (tid < 256) blds[tid] = Wb[tid];
  __syncthreads();

  float* OutLds = (float*)Alds;          // reused after barrier

  const int mTile  = wave & 1;           // 2 M tiles (32 rows)
  const int nBase0 = (wave >> 1) * 2;    // 2 N tiles per wave (16 total)
  const int nTiles = NUM_NODES / 32;     // 625

  for (int tile = blockIdx.x; tile < nTiles; tile += gridDim.x) {
    const int rowG = tile * 32;

    // load concat tile (already f16) -> LDS
    for (int it = 0; it < 4; ++it) {
      int flat = it * 512 + tid;          // 2048 x 8-f16 chunks
      int row  = flat >> 6;
      int c8   = (flat & 63) << 3;
      *(v8h*)&Alds[row * WP + c8] =
          *(const v8h*)(concatIn + (long)(rowG + row) * CAT_F + c8);
    }
    __syncthreads();

    // D = A(32x512) x Ww^T(512x256): per wave 2 tiles, K-loop of 16
    v8f acc[2] = {};
    const int arow = (mTile * 16 + laneN) * WP;
    #pragma unroll
    for (int kb = 0; kb < 512; kb += 32) {
      v8h alo = *(const v8h*)&Alds[arow + kb + laneH * 16];
      v8h ahi = *(const v8h*)&Alds[arow + kb + laneH * 16 + 8];
      v16h a  = cat8(alo, ahi);
      #pragma unroll
      for (int j = 0; j < 2; ++j) {
        const v16h b = *(const v16h*)&Wlds[((nBase0 + j) * 16 + laneN) * WP + kb + laneH * 16];
        acc[j] = __builtin_amdgcn_wmma_f32_16x16x32_f16(
                     false, a, false, b, (short)0, acc[j], false, false);
      }
    }
    __syncthreads();   // everyone done reading Alds before aliasing as OutLds

    // bias + ReLU into f32 LDS tile
    const int rowBase = mTile * 16 + laneH * 8;
    #pragma unroll
    for (int j = 0; j < 2; ++j) {
      int col  = (nBase0 + j) * 16 + laneN;
      float bb = blds[col];
      #pragma unroll
      for (int r = 0; r < 8; ++r) {
        float v = fmaxf(acc[j][r] + bb, 0.f);
        OutLds[(rowBase + r) * 256 + col] = v;
      }
    }
    __syncthreads();

    // L2 normalize: each wave owns 2 rows
    #pragma unroll
    for (int rr = 0; rr < 2; ++rr) {
      int row = wave * 2 + rr;
      float s = 0.f;
      #pragma unroll
      for (int c = lane; c < 256; c += 32) {
        float v = OutLds[row * 256 + c];
        s += v * v;
      }
      for (int off = 16; off > 0; off >>= 1) s += __shfl_xor(s, off, 32);
      float rinv = rsqrtf(s);
      #pragma unroll
      for (int c = lane; c < 256; c += 32)
        out[(long)(rowG + row) * OUT_F + c] = OutLds[row * 256 + c] * rinv;
    }
    __syncthreads();   // protect OutLds/Alds before next tile load
  }
}

extern "C" void kernel_launch(void* const* d_in, const int* in_sizes, int n_in,
                              void* d_out, int out_size, void* d_ws, size_t ws_size,
                              hipStream_t stream) {
  const float* h          = (const float*)d_in[0];
  const int*   nodeset    = (const int*)  d_in[1];
  const int*   nb_nodes   = (const int*)  d_in[2];
  const float* nb_weights = (const float*)d_in[3];
  const float* Qw         = (const float*)d_in[4];
  const float* Qb         = (const float*)d_in[5];
  const float* Ww         = (const float*)d_in[6];
  const float* Wb         = (const float*)d_in[7];

  _Float16* concatBuf = (_Float16*)d_ws;   // [20000][512] f16 = 20.48 MB
  float*    out       = (float*)d_out;

  k1_aggregate<<<2000, 512, 0, stream>>>(h, nodeset, nb_nodes, nb_weights,
                                         Qw, Qb, concatBuf);
  k2_output<<<160, 512, 0, stream>>>(concatBuf, Ww, Wb, out);
}
